// KANLayer_89962384982633
// MI455X (gfx1250) — matrix operand order repaired and checked by reading, not compiled, not run on hardware
//
#include <hip/hip_runtime.h>
#include <stdint.h>

typedef __attribute__((ext_vector_type(16))) _Float16 v16h;
typedef __attribute__((ext_vector_type(8)))  float    v8f;
typedef __attribute__((ext_vector_type(4)))  int      v4i;

typedef __attribute__((address_space(1))) v4i gv4i;   // global (HIP "__device__")
typedef __attribute__((address_space(3))) v4i lv4i;   // LDS    (HIP "__shared__")

#define TB     256
#define NWAVES 8

#if __has_builtin(__builtin_amdgcn_global_load_async_to_lds_b128)
#define HAVE_ASYNC_LDS 1
#else
#define HAVE_ASYNC_LDS 0
#endif

__device__ __forceinline__ float hw_tanh(float v) {
#if __has_builtin(__builtin_amdgcn_tanhf)
  return __builtin_amdgcn_tanhf(v);   // gfx1250 v_tanh_f32
#else
  return tanhf(v);
#endif
}

__device__ __forceinline__ void wait_async_all() {
#if __has_builtin(__builtin_amdgcn_s_wait_asynccnt)
  __builtin_amdgcn_s_wait_asynccnt(0);
#else
  asm volatile("s_wait_asynccnt 0" ::: "memory");
#endif
}

// out[b] = sum_{d,h} tanh(x[b,d]*w1[d,h]+b1[d,h]) * w2[d,h] + sum(b2)
// B=65536, D=256, H=16. One wave owns 16 b-rows; the 4096-long flattened
// (d,h) contraction runs on v_wmma_f32_16x16x32_f16 (K=32 chunks, column 0
// of the B operand carries the f16 w2 chunk).
__global__ __launch_bounds__(TB)
void kan_wmma_kernel(const float* __restrict__ x,
                     const float* __restrict__ w1,
                     const float* __restrict__ b1,
                     const float* __restrict__ w2,
                     const float* __restrict__ b2,
                     float* __restrict__ out)
{
  __shared__ float    lw1[4096];              // w1[d*16+h], f32
  __shared__ float    lb1[4096];              // b1[d*16+h], f32
  __shared__ uint32_t lw2z[8 + 2048];         // [0..7]=zeros, then w2 f16x2 pairs
  __shared__ float    xt[NWAVES][16][64];     // per-wave x tile [16 b-rows][64 d]
  __shared__ float    sumb2_sh;

  const int t = threadIdx.x;

  // ---- stage params into LDS (block-wide, coalesced) ----
  {
    const float4* w1v = (const float4*)w1;
    const float4* b1v = (const float4*)b1;
    float4* lw1v = (float4*)lw1;
    float4* lb1v = (float4*)lb1;
    #pragma unroll
    for (int i = t; i < 1024; i += TB) { lw1v[i] = w1v[i]; lb1v[i] = b1v[i]; }

    const float2* w2v = (const float2*)w2;
    #pragma unroll
    for (int i = t; i < 2048; i += TB) {
      float2 v = w2v[i];
      union { _Float16 h[2]; uint32_t u; } pk;
      pk.h[0] = (_Float16)v.x;
      pk.h[1] = (_Float16)v.y;
      lw2z[8 + i] = pk.u;
    }
    if (t < 8) lw2z[t] = 0u;   // zero region read by non-column-0 lanes
    if (t == 0) {
      float s = 0.f;
      for (int d = 0; d < 256; ++d) s += b2[d];
      sumb2_sh = s;
    }
  }
  __syncthreads();

  const int wave = t >> 5;
  const int lane = t & 31;
  const int m    = lane & 15;   // A-matrix row owned by this lane
  const int hsel = lane >> 4;   // K-half selector per ISA 7.12.2 layout
  const int bbase = (blockIdx.x * NWAVES + wave) * 16;

  // Column-0 lanes (0 and 16) read real w2 pairs; all others read the
  // 32B zero region at lw2z[0] -> branch-free, no per-word masking.
  const uint32_t idxmask = (m == 0) ? 0xFFFFFFFFu : 0u;

  v8f acc = {};  // f32 C/D fragment; we use column N=0

  for (int dt = 0; dt < 4; ++dt) {
    const int dbase = dt * 64;

    // prefetch next x tile toward caches (global_prefetch_b8)
    if (dt < 3) {
      const float* nxt = &x[(bbase + m) * 256 + dbase + 64 + hsel * 32];
      __builtin_prefetch(nxt, 0, 0);
    }

    // ---- stage x tile [16 rows x 64 d] into LDS ----
#if HAVE_ASYNC_LDS
    {
      #pragma unroll
      for (int i = lane; i < 256; i += 32) {
        const int row = i >> 4, c4 = i & 15;
        const float4* gp = ((const float4*)x) + (bbase + row) * 64 + (dbase >> 2) + c4;
        float4*       lp = ((float4*)&xt[wave][row][0]) + c4;
        __builtin_amdgcn_global_load_async_to_lds_b128(
            (gv4i*)gp, (lv4i*)lp, /*offset=*/0, /*cpol=*/0);
      }
      wait_async_all();
      asm volatile("" ::: "memory");
    }
#else
    {
      const float4* xr = (const float4*)x;
      #pragma unroll
      for (int i = lane; i < 256; i += 32) {
        const int row = i >> 4, c4 = i & 15;
        float4 v = xr[(bbase + row) * 64 + (dbase >> 2) + c4];
        ((float4*)&xt[wave][row][0])[c4] = v;
      }
    }
#endif

    // ---- 32 chunks of K=32 (each chunk = 2 d-values x 16 h) ----
    for (int c = 0; c < 32; ++c) {
      const int   d0   = dbase + 2 * c;
      const float xv0  = xt[wave][m][2 * c];
      const float xv1  = xt[wave][m][2 * c + 1];
      const float* w1r0 = &lw1[d0 * 16];
      const float* b1r0 = &lb1[d0 * 16];
      const float* w1r1 = w1r0 + 16;
      const float* b1r1 = b1r0 + 16;

      // A fragment per ISA 7.12.2 (16-bit A 16x32, wave32).
      v16h a;
      #pragma unroll
      for (int j = 0; j < 8; ++j) {
        const bool lo = (j < 4);
        const float  xv  = lo ? xv0  : xv1;
        const float* w1r = lo ? w1r0 : w1r1;
        const float* b1r = lo ? b1r0 : b1r1;
        #pragma unroll
        for (int p = 0; p < 2; ++p) {
          const int hh = 2 * (j & 3) + p + 8 * hsel;
          const float v = hw_tanh(fmaf(xv, w1r[hh], b1r[hh]));
          a[2 * j + p] = (_Float16)v;
        }
      }

      // B fragment: 2x ds_load_b128 straight into the WMMA operand.
      // Active lanes (0,16) index their w2 chunk; others hit the zero
      // prefix for q0 (q1 of those lanes only feeds discarded columns). Both addresses are 16B-aligned; LDS broadcasts.
      const int ibase = (dt * 32 + c) * 16 + hsel * 4;        // f16-pair index
      const uint32_t idx = (uint32_t)(8 + ibase) & idxmask;   // 0 for inactive
      const uint4* w2q = (const uint4*)&lw2z[idx];
      uint4 q0 = w2q[0];
      uint4 q1 = w2q[2];
      union { v16h v; uint32_t u[8]; } bf;
      bf.u[0] = q0.x; bf.u[1] = q0.y; bf.u[2] = q0.z; bf.u[3] = q0.w;
      bf.u[4] = q1.x; bf.u[5] = q1.y; bf.u[6] = q1.z; bf.u[7] = q1.w;

      acc = __builtin_amdgcn_wmma_f32_16x16x32_f16(
          /*neg_a=*/false, a, /*neg_b=*/false, bf.v,
          /*c_mod=*/(short)0, acc, /*reuse_a=*/false, /*reuse_b=*/false);
    }
  }

  // Column N=0 of C/D: lane 0 holds M=0..7, lane 16 holds M=8..15.
  if (m == 0) {
    const float sb = sumb2_sh;
    #pragma unroll
    for (int r = 0; r < 8; ++r)
      out[bbase + hsel * 8 + r] = acc[r] + sb;
  }
}

extern "C" void kernel_launch(void* const* d_in, const int* in_sizes, int n_in,
                              void* d_out, int out_size, void* d_ws, size_t ws_size,
                              hipStream_t stream) {
  const float* x  = (const float*)d_in[0];
  const float* w1 = (const float*)d_in[1];
  const float* b1 = (const float*)d_in[2];
  const float* w2 = (const float*)d_in[3];
  const float* b2 = (const float*)d_in[4];
  float* out = (float*)d_out;

  // 4096 b-blocks of 16 rows, one per wave; 8 waves per 256-thread block.
  dim3 grid(512), block(TB);
  hipLaunchKernelGGL(kan_wmma_kernel, grid, block, 0, stream,
                     x, w1, b1, w2, b2, out);
}
